// AttentionHead_63058709840586
// MI455X (gfx1250) — matrix operand order repaired and checked by reading, not compiled, not run on hardware
//
#include <hip/hip_runtime.h>
#include <math.h>

typedef __attribute__((ext_vector_type(16))) __bf16 v16bf;
typedef __attribute__((ext_vector_type(8)))  __bf16 bf8;
typedef __attribute__((ext_vector_type(4)))  __bf16 bf4;
typedef __attribute__((ext_vector_type(8)))  float  v8f;
typedef __attribute__((ext_vector_type(4)))  int    i4;
typedef __attribute__((address_space(1)))    i4     gi4;   // global int4
typedef __attribute__((address_space(3)))    i4     li4;   // LDS int4

#define BATCH 8
#define SEQ   2048
#define IN    512
#define HID   512
#define XSB   56      /* bf16 LDS stride: 112B rows, 16B aligned, spreads banks */
#define PS    264     /* bf16 LDS stride for P tile: 528B rows, 16B aligned    */
#define MS    260     /* int32 LDS stride for mask tile: 1040B rows, 16B align */

// Async global->LDS mask staging (CDNA5 ASYNCcnt path), guarded so the file
// still compiles on toolchains lacking the builtins.
#if defined(__has_builtin)
#  if __has_builtin(__builtin_amdgcn_global_load_async_to_lds_b128) && \
      __has_builtin(__builtin_amdgcn_s_wait_asynccnt)
#    define USE_ASYNC_MASK 1
#  endif
#endif
#ifndef USE_ASYNC_MASK
#  define USE_ASYNC_MASK 0
#endif

// A fragment (ISA 16-bit A layout): lane ln = row M; half h selects
// K runs {h*8 .. h*8+7} and {16+h*8 .. 16+h*8+7} -> two contiguous b128 loads.
template <typename P>
static __device__ inline v16bf load_a_frag(const P* rowp, int h)
{
    bf8 lo = *(const bf8*)(rowp + (h << 3));
    bf8 hi = *(const bf8*)(rowp + 16 + (h << 3));
    v16bf r;
    #pragma unroll
    for (int i = 0; i < 8; ++i) { r[i] = lo[i]; r[i + 8] = hi[i]; }
    return r;
}

// B fragment (32x16): lane ln = col N; half h selects K runs {h*8 .. h*8+7}
// and {16+h*8 .. 16+h*8+7} (same as A) -> two b128 loads.
template <typename P>
static __device__ inline v16bf load_b_frag(const P* rowp, int h)
{
    bf8 lo = *(const bf8*)(rowp + (h << 3));
    bf8 hi = *(const bf8*)(rowp + 16 + (h << 3));
    v16bf r;
    #pragma unroll
    for (int i = 0; i < 8; ++i) { r[i] = lo[i]; r[i + 8] = hi[i]; }
    return r;
}

static __device__ inline void store_bf4(__bf16* dst, float4 v)
{
    bf4 t;
    t[0] = (__bf16)v.x; t[1] = (__bf16)v.y;
    t[2] = (__bf16)v.z; t[3] = (__bf16)v.w;
    *(bf4*)dst = t;                       // 8B LDS store (cvt_pk at staging)
}

// ---------------------------------------------------------------------------
// Kernel 1: Y = X * W^T + b for q/k/v (blockIdx.z selects). bf16 outputs.
// Q,K row-major (token, hid); V transposed (hid, token).
// Block 256 thr (8 waves); tile 128 rows x 64 cols; K staged 32-wide in LDS
// as bf16; global fetch of chunk i+1 overlaps WMMAs of chunk i.
// ---------------------------------------------------------------------------
__global__ __launch_bounds__(256) void qkv_proj_kernel(
    const float* __restrict__ X,
    const float* __restrict__ Wq, const float* __restrict__ bq,
    const float* __restrict__ Wk, const float* __restrict__ bk,
    const float* __restrict__ Wv, const float* __restrict__ bv,
    __bf16* __restrict__ Q, __bf16* __restrict__ K, __bf16* __restrict__ VT)
{
    const int z = blockIdx.z;
    const float* __restrict__ W    = (z == 0) ? Wq : (z == 1) ? Wk : Wv;
    const float* __restrict__ bias = (z == 0) ? bq : (z == 1) ? bk : bv;

    __shared__ __bf16 Xt[128 * XSB];   // 14.0 KB
    __shared__ __bf16 Wt[64 * XSB];    //  7.0 KB

    const int tid  = threadIdx.x;
    const int wave = tid >> 5;
    const int lane = tid & 31;
    const int h    = lane >> 4;
    const int ln   = lane & 15;

    const int r0 = blockIdx.y * 128;
    const int c0 = blockIdx.x * 64;

    v8f acc[4] = {};

    float4 xr[4], wr[2];
    #pragma unroll
    for (int i = 0; i < 4; ++i) {
        int li = i * 256 + tid, row = li >> 3, kq = (li & 7) << 2;
        xr[i] = *(const float4*)(X + (size_t)(r0 + row) * IN + kq);
    }
    #pragma unroll
    for (int i = 0; i < 2; ++i) {
        int li = i * 256 + tid, row = li >> 3, kq = (li & 7) << 2;
        wr[i] = *(const float4*)(W + (size_t)(c0 + row) * IN + kq);
    }

    for (int k0 = 0; k0 < IN; k0 += 32) {
        __syncthreads();   // previous compute done reading LDS
        #pragma unroll
        for (int i = 0; i < 4; ++i) {
            int li = i * 256 + tid, row = li >> 3, kq = (li & 7) << 2;
            store_bf4(Xt + row * XSB + kq, xr[i]);
        }
        #pragma unroll
        for (int i = 0; i < 2; ++i) {
            int li = i * 256 + tid, row = li >> 3, kq = (li & 7) << 2;
            store_bf4(Wt + row * XSB + kq, wr[i]);
        }
        __syncthreads();   // LDS ready

        if (k0 + 32 < IN) {  // overlap next global fetch with WMMAs
            #pragma unroll
            for (int i = 0; i < 4; ++i) {
                int li = i * 256 + tid, row = li >> 3, kq = (li & 7) << 2;
                xr[i] = *(const float4*)(X + (size_t)(r0 + row) * IN + k0 + 32 + kq);
            }
            #pragma unroll
            for (int i = 0; i < 2; ++i) {
                int li = i * 256 + tid, row = li >> 3, kq = (li & 7) << 2;
                wr[i] = *(const float4*)(W + (size_t)(c0 + row) * IN + k0 + 32 + kq);
            }
        }

        // load A and all 4 B fragments first (one ds clause), then 4 WMMAs
        v16bf a = load_a_frag(Xt + (wave * 16 + ln) * XSB, h);
        v16bf bfr[4];
        #pragma unroll
        for (int c = 0; c < 4; ++c)
            bfr[c] = load_b_frag(Wt + (c * 16 + ln) * XSB, h);
        #pragma unroll
        for (int c = 0; c < 4; ++c)
            acc[c] = __builtin_amdgcn_wmma_f32_16x16x32_bf16(
                false, a, false, bfr[c], (short)0, acc[c], false, false);
    }

    // epilogue: +bias, cvt bf16, store. C layout: VGPR r <-> row r+8h
    #pragma unroll
    for (int c = 0; c < 4; ++c) {
        int n = c0 + c * 16 + ln;
        float bval = bias[n];
        #pragma unroll
        for (int r = 0; r < 8; ++r) {
            int row = r0 + wave * 16 + r + (h << 3);
            __bf16 o = (__bf16)(acc[c][r] + bval);
            if (z == 2) {
                int bb = row >> 11, s = row & (SEQ - 1);
                VT[((size_t)bb * HID + n) * SEQ + s] = o;
            } else if (z == 0) {
                Q[(size_t)row * HID + n] = o;
            } else {
                K[(size_t)row * HID + n] = o;
            }
        }
    }
}

// ---------------------------------------------------------------------------
// Kernel 2: flash attention. Block = (16-query tile, batch), 8 waves.
// 256-key blocks: async-stage this wave's 16x32 mask subtile into LDS while
// the 32 score WMMAs run; online softmax via LDS-shared row stats; P (bf16)
// in LDS; each wave does PV for its 64-wide hidden slice. f32 output.
// ---------------------------------------------------------------------------
__global__ __launch_bounds__(256) void attn_kernel(
    const __bf16* __restrict__ Q, const __bf16* __restrict__ K,
    const __bf16* __restrict__ VT, const int* __restrict__ mask,
    float* __restrict__ out)
{
    __shared__ __bf16 Pt[16 * PS];     // 8.25 KB
    __shared__ int    Mi[16 * MS];     // 16.6 KB mask tile (async staged)
    __shared__ float  wmax[8][16];
    __shared__ float  wsum[8][16];
    __shared__ float  mrow[16], lrow[16], arow[16];

    const int tid  = threadIdx.x;
    const int wave = tid >> 5;
    const int lane = tid & 31;
    const int h    = lane >> 4;
    const int ln   = lane & 15;

    const int q0 = blockIdx.x * 16;
    const int b  = blockIdx.y;

    if (tid < 16) { mrow[tid] = -3.0e38f; lrow[tid] = 0.0f; }
    __syncthreads();

    // resident Q: 16 rows x 512 = 16 A-fragments (128 VGPRs; wave32 has 1024)
    v16bf qa[16];
    {
        const __bf16* qrow = Q + ((size_t)b * SEQ + q0 + ln) * HID;
        #pragma unroll
        for (int kt = 0; kt < 16; ++kt)
            qa[kt] = load_a_frag(qrow + kt * 32, h);
    }

    v8f oacc[4] = {};
    const float inv_sqrt = 0.044194173824159216f;  // 1/sqrt(512)
    const float NEGBIG   = -2147483648.0f;         // reference NEG_INF
    const int*  mp       = mask + ((size_t)b * SEQ + q0) * SEQ;

    for (int kb = 0; kb < SEQ; kb += 256) {
        const int key0 = kb + wave * 32;

#if USE_ASYNC_MASK
        // async-stage this wave's 16x32 mask subtile; overlaps the QK^T WMMAs.
        // Each wave reads only what it wrote -> only s_wait_asynccnt needed;
        // LDS region reuse across iterations is ordered by the end barrier.
        #pragma unroll
        for (int i = 0; i < 4; ++i) {
            int ci  = i * 32 + lane;            // 0..127
            int row = ci >> 3;                  // 16 rows
            int c4  = (ci & 7) << 2;            // 8 chunks of 4 ints
            const int* g = mp + (size_t)row * SEQ + kb + wave * 32 + c4;
            int*       l = Mi + row * MS + wave * 32 + c4;
            __builtin_amdgcn_global_load_async_to_lds_b128(
                (gi4*)(void*)g, (li4*)(void*)l, 0, 0);
        }
#endif
        // hint next key block into cache while we compute on this one
        if (kb + 256 < SEQ) {
            __builtin_prefetch(K + ((size_t)b * SEQ + key0 + 256 + lane) * HID, 0, 0);
            __builtin_prefetch(VT + ((size_t)b * HID + wave * 64 + lane) * SEQ
                                  + kb + 256, 0, 0);
        }

        // ---- scores: 16 queries x 32 keys (this wave's slice) ----
        v8f s0 = {}, s1 = {};
        const __bf16* kbase = K + ((size_t)b * SEQ + key0 + ln) * HID;
        #pragma unroll
        for (int kt = 0; kt < 16; ++kt) {
            v16bf b0 = load_b_frag(kbase + kt * 32, h);
            v16bf b1 = load_b_frag(kbase + (size_t)16 * HID + kt * 32, h);
            s0 = __builtin_amdgcn_wmma_f32_16x16x32_bf16(false, qa[kt], false, b0,
                                                         (short)0, s0, false, false);
            s1 = __builtin_amdgcn_wmma_f32_16x16x32_bf16(false, qa[kt], false, b1,
                                                         (short)0, s1, false, false);
        }

        // ---- scale + mask ----
#if USE_ASYNC_MASK
        __builtin_amdgcn_s_wait_asynccnt(0);
        #pragma unroll
        for (int r = 0; r < 8; ++r) {
            int m   = r + (h << 3);
            int mv0 = Mi[m * MS + wave * 32 + ln];
            int mv1 = Mi[m * MS + wave * 32 + 16 + ln];
            s0[r] = mv0 ? s0[r] * inv_sqrt : NEGBIG;
            s1[r] = mv1 ? s1[r] * inv_sqrt : NEGBIG;
        }
#else
        #pragma unroll
        for (int r = 0; r < 8; ++r) {
            int m   = r + (h << 3);
            int key = key0 + ln;
            int mv0 = mp[(size_t)m * SEQ + key];
            int mv1 = mp[(size_t)m * SEQ + key + 16];
            s0[r] = mv0 ? s0[r] * inv_sqrt : NEGBIG;
            s1[r] = mv1 ? s1[r] * inv_sqrt : NEGBIG;
        }
#endif

        // ---- per-wave row max (butterfly within 16-lane half) ----
        #pragma unroll
        for (int r = 0; r < 8; ++r) {
            float mx = fmaxf(s0[r], s1[r]);
            mx = fmaxf(mx, __shfl_xor(mx, 1, 32));
            mx = fmaxf(mx, __shfl_xor(mx, 2, 32));
            mx = fmaxf(mx, __shfl_xor(mx, 4, 32));
            mx = fmaxf(mx, __shfl_xor(mx, 8, 32));
            if (ln == 0) wmax[wave][r + (h << 3)] = mx;
        }
        __syncthreads();

        if (wave == 0 && lane < 16) {
            float bm = wmax[0][lane];
            #pragma unroll
            for (int w = 1; w < 8; ++w) bm = fmaxf(bm, wmax[w][lane]);
            float mo = mrow[lane];
            float mn = fmaxf(mo, bm);
            arow[lane] = __expf(mo - mn);
            mrow[lane] = mn;
        }
        __syncthreads();

        // ---- P = exp(s - m_new) -> bf16 LDS; per-wave row sums ----
        #pragma unroll
        for (int r = 0; r < 8; ++r) {
            int   m  = r + (h << 3);
            float mn = mrow[m];
            float p0 = __expf(s0[r] - mn);
            float p1 = __expf(s1[r] - mn);
            Pt[m * PS + wave * 32 + ln]      = (__bf16)p0;
            Pt[m * PS + wave * 32 + 16 + ln] = (__bf16)p1;
            float sm = p0 + p1;
            sm += __shfl_xor(sm, 1, 32);
            sm += __shfl_xor(sm, 2, 32);
            sm += __shfl_xor(sm, 4, 32);
            sm += __shfl_xor(sm, 8, 32);
            if (ln == 0) wsum[wave][m] = sm;
        }
        __syncthreads();

        if (wave == 0 && lane < 16) {
            float bs = 0.0f;
            #pragma unroll
            for (int w = 0; w < 8; ++w) bs += wsum[w][lane];
            lrow[lane] = lrow[lane] * arow[lane] + bs;
        }

        // ---- rescale O, then PV over this 256-key block ----
        float al[8];
        #pragma unroll
        for (int r = 0; r < 8; ++r) al[r] = arow[r + (h << 3)];
        #pragma unroll
        for (int c = 0; c < 4; ++c)
            #pragma unroll
            for (int r = 0; r < 8; ++r) oacc[c][r] *= al[r];

        #pragma unroll
        for (int kt = 0; kt < 8; ++kt) {
            v16bf pa = load_a_frag(Pt + ln * PS + kt * 32, h);
            #pragma unroll
            for (int c = 0; c < 4; ++c) {
                int n = wave * 64 + c * 16 + ln;
                v16bf vb = load_b_frag(VT + ((size_t)b * HID + n) * SEQ
                                          + kb + kt * 32, h);
                oacc[c] = __builtin_amdgcn_wmma_f32_16x16x32_bf16(
                    false, pa, false, vb, (short)0, oacc[c], false, false);
            }
        }
        __syncthreads();   // protect Pt/Mi/arow/wmax/wsum before next iteration
    }

    // ---- finalize: O / l, f32 store ----
    float li[8];
    #pragma unroll
    for (int r = 0; r < 8; ++r) li[r] = 1.0f / lrow[r + (h << 3)];
    #pragma unroll
    for (int c = 0; c < 4; ++c) {
        int n = wave * 64 + c * 16 + ln;
        #pragma unroll
        for (int r = 0; r < 8; ++r) {
            int row = q0 + r + (h << 3);
            out[((size_t)b * SEQ + row) * HID + n] = oacc[c][r] * li[r];
        }
    }
}

// ---------------------------------------------------------------------------
extern "C" void kernel_launch(void* const* d_in, const int* in_sizes, int n_in,
                              void* d_out, int out_size, void* d_ws, size_t ws_size,
                              hipStream_t stream)
{
    const float* X    = (const float*)d_in[0];
    const int*   mask = (const int*)d_in[1];
    const float* Wq   = (const float*)d_in[2];
    const float* bq   = (const float*)d_in[3];
    const float* Wk   = (const float*)d_in[4];
    const float* bk   = (const float*)d_in[5];
    const float* Wv   = (const float*)d_in[6];
    const float* bv   = (const float*)d_in[7];

    const size_t nElem = (size_t)BATCH * SEQ * HID;   // 8 Mi elems
    __bf16* Q  = (__bf16*)d_ws;
    __bf16* K  = Q + nElem;
    __bf16* VT = K + nElem;

    qkv_proj_kernel<<<dim3(HID / 64, (BATCH * SEQ) / 128, 3), 256, 0, stream>>>(
        X, Wq, bq, Wk, bk, Wv, bv, Q, K, VT);

    attn_kernel<<<dim3(SEQ / 16, BATCH), 256, 0, stream>>>(
        Q, K, VT, mask, (float*)d_out);
}
